// ContrasPQ_23029614641839
// MI455X (gfx1250) — matrix operand order repaired and checked by reading, not compiled, not run on hardware
//
#include <hip/hip_runtime.h>

// CDNA5 / gfx1250 PQ hard-assignment kernel, round 2.
// out[b, p*8:(p+1)*8] = codebook[p, argmin_k(||c_k||^2 - 2 v.c_k), :]
//
// One V_WMMA_F32_16X16X32_F16 per 16x16 score tile computes the FULL score:
// the 32-deep K dim is packed as [ah.ch | al.ch | ah.cl | 1*cnh,1*cnl,0...]
// where x = xh + xl is an f16 hi/lo split (=> ~2^-22 relative accuracy) and
// cn = ||c||^2 is folded in via constant-1 A slots. C operand is zero.
// 16-bit WMMA is TRANS-tracked on CDNA5 -> co-executes with the argmin VALU.

typedef _Float16 half_t;
typedef __attribute__((ext_vector_type(16))) _Float16 v16h;
typedef __attribute__((ext_vector_type(8)))  float    v8f;

constexpr int kB   = 8192;
constexpr int kEMB = 768;
constexpr int kP   = 96;
constexpr int kK   = 256;
constexpr int kD   = 8;
constexpr int kRowsPerBlock = 128;   // 8 waves * 16 rows

__global__ __launch_bounds__(256, 2)
void ContrasPQ_assign_kernel(const float* __restrict__ vecs,
                             const float* __restrict__ codebook,
                             float* __restrict__ out)
{
    // B-fragment rows, 32 bytes per codeword, one ds_load_b128 x2 per tile:
    //   s_blo[k] = { ch[0..7], ch[0..7] }              (lanes 0-15: K 0-15)
    //   s_bhi[k] = { cl[0..7], cnh, cnl, 0 x6 }        (lanes 16-31: K 16-31)
    __shared__ unsigned s_blo[kK * 8];   // 8 KB
    __shared__ unsigned s_bhi[kK * 8];   // 8 KB
    __shared__ float    s_cbf[kK * kD];  // 8 KB, exact f32 rows for the gather

    const int tid = threadIdx.x;
    const int p   = blockIdx.y;

    // ---- Stage codebook[p]: one codeword per thread --------------------------
    {
        const float4* src = (const float4*)(codebook + (size_t)p * kK * kD);
        float4 c0 = src[tid * 2 + 0];
        float4 c1 = src[tid * 2 + 1];
        ((float4*)s_cbf)[tid * 2 + 0] = c0;
        ((float4*)s_cbf)[tid * 2 + 1] = c1;

        float cf[8] = {c0.x, c0.y, c0.z, c0.w, c1.x, c1.y, c1.z, c1.w};
        half_t ch[8], cl[8];
        float cn = 0.0f;
#pragma unroll
        for (int i = 0; i < 8; ++i) {
            ch[i] = (half_t)cf[i];
            cl[i] = (half_t)(cf[i] - (float)ch[i]);
            cn += cf[i] * cf[i];
        }
        half_t cnh = (half_t)cn;
        half_t cnl = (half_t)(cn - (float)cnh);

        union { half_t h[16]; uint4 u[2]; } lo, hi;
#pragma unroll
        for (int i = 0; i < 8; ++i) {
            lo.h[i]     = ch[i];
            lo.h[8 + i] = ch[i];
            hi.h[i]     = cl[i];
        }
        hi.h[8] = cnh;
        hi.h[9] = cnl;
#pragma unroll
        for (int i = 10; i < 16; ++i) hi.h[i] = (half_t)0.0f;

        ((uint4*)s_blo)[tid * 2 + 0] = lo.u[0];
        ((uint4*)s_blo)[tid * 2 + 1] = lo.u[1];
        ((uint4*)s_bhi)[tid * 2 + 0] = hi.u[0];
        ((uint4*)s_bhi)[tid * 2 + 1] = hi.u[1];
    }
    __syncthreads();

    const int wave  = tid >> 5;
    const int lane  = tid & 31;
    const int lhalf = lane >> 4;      // A: K-range selector; B: K-range selector
    const int lrow  = lane & 15;      // A: M row; B/C: N column
    const int b0    = blockIdx.x * kRowsPerBlock + wave * 16;

    // ---- A fragment: hi/lo split of (-2 * v_row), row M = lane&15 ------------
    // half 0: [ah x8 | al x8]   (K 0-7: ah.ch, K 8-15: al.ch)
    // half 1: [ah x8 | 1,1,0..] (K 16-23: ah.cl, K 24-25: 1*cnh, 1*cnl)
    const float* vrow = vecs + (size_t)(b0 + lrow) * kEMB + p * kD;
    float4 va = *(const float4*)(vrow);
    float4 vb = *(const float4*)(vrow + 4);
    float av[8] = {-2.0f * va.x, -2.0f * va.y, -2.0f * va.z, -2.0f * va.w,
                   -2.0f * vb.x, -2.0f * vb.y, -2.0f * vb.z, -2.0f * vb.w};
    v16h A;
#pragma unroll
    for (int i = 0; i < 8; ++i) {
        half_t ah = (half_t)av[i];
        half_t al = (half_t)(av[i] - (float)ah);
        half_t ones = (half_t)((i < 2) ? 1.0f : 0.0f);
        A[i]     = ah;
        A[8 + i] = lhalf ? ones : al;
    }

    float minv[8];
    int   mink[8];
#pragma unroll
    for (int r = 0; r < 8; ++r) { minv[r] = 3.4e38f; mink[r] = 0; }

    const unsigned* bbase = lhalf ? s_bhi : s_blo;

    // ---- 16 column tiles; WMMA emits score = ||c||^2 - 2 v.c directly --------
#pragma unroll 4
    for (int t = 0; t < 16; ++t) {
        const int col = t * 16 + lrow;
        v16h Bm = *(const v16h*)(const void*)(bbase + col * 8);   // 32B LDS row
        v8f acc = {};                                             // C = 0
        acc = __builtin_amdgcn_wmma_f32_16x16x32_f16(false, A, false, Bm,
                                                     (short)0, acc, false, false);
        // acc[r] = score for row (8*lhalf + r), column col
#pragma unroll
        for (int r = 0; r < 8; ++r) {
            float s  = acc[r];
            bool  lt = s < minv[r];
            minv[r] = lt ? s   : minv[r];
            mink[r] = lt ? col : mink[r];
        }
    }

    // ---- Butterfly argmin across the 16 N-columns (stays within each half) ---
#pragma unroll
    for (int off = 1; off < 16; off <<= 1) {
#pragma unroll
        for (int r = 0; r < 8; ++r) {
            float ov = __shfl_xor(minv[r], off);
            int   ok = __shfl_xor(mink[r], off);
            if (ov < minv[r] || (ov == minv[r] && ok < mink[r])) {
                minv[r] = ov;
                mink[r] = ok;
            }
        }
    }
    // Every lane in half h now holds argmin for rows 8h..8h+7.

    // ---- Gather winning codeword (exact f32), coalesced float4 stores --------
    const int myrow = lane >> 1;       // lanes 0-15 -> rows 0-7 ; 16-31 -> 8-15
    const int j     = myrow & 7;
    int kwin = mink[0];
#pragma unroll
    for (int r = 1; r < 8; ++r) kwin = (j == r) ? mink[r] : kwin;

    float4 c4 = *(const float4*)(s_cbf + kwin * kD + (lane & 1) * 4);
    float* orow = out + (size_t)(b0 + myrow) * kEMB + p * kD + (lane & 1) * 4;
    *(float4*)orow = c4;
}

extern "C" void kernel_launch(void* const* d_in, const int* in_sizes, int n_in,
                              void* d_out, int out_size, void* d_ws, size_t ws_size,
                              hipStream_t stream) {
    const float* vecs     = (const float*)d_in[0];   // [8192, 768] f32
    const float* codebook = (const float*)d_in[1];   // [96, 256, 8] f32
    float* out            = (float*)d_out;           // [8192, 768] f32

    dim3 grid(kB / kRowsPerBlock, kP);   // (64, 96)
    ContrasPQ_assign_kernel<<<grid, 256, 0, stream>>>(vecs, codebook, out);
}
